// AtomProteinAttentionGNN_82076825027006
// MI455X (gfx1250) — compile-verified, matrix operand order checked by reading, not run on hardware
//
#include <hip/hip_runtime.h>
#include <hip/hip_bf16.h>

// ---------------- types ----------------
typedef __attribute__((ext_vector_type(16))) __bf16        v16bf;
typedef __attribute__((ext_vector_type(8)))  float         v8f;
typedef __attribute__((ext_vector_type(8)))  unsigned short v8u;
typedef __attribute__((ext_vector_type(16))) unsigned short v16u;

// ---------------- model constants ----------------
#define Bc    48
#define NMAXc 64
#define Lc    512
#define NFc   74
#define EFc   12
#define PFc   256
#define Gc    200
#define Hc    4
#define Dc    50
#define Vc    (Bc * NMAXc)          // 3072
#define G3c   (3 * Gc)              // 600
#define Dpc   64                    // D padded to 32-multiple (K dim of attn GEMMs)
#define Gpc   224                   // G padded to 32-multiple
#define KNFp  96                    // 74 (and 86) padded
#define RPc   (Bc * Lc)             // 24576 protein rows
#define RTc   (Bc * NMAXc * NMAXc)  // 196608 pair rows
#define NP200 224                   // N=200 padded to 32 (for 32x32 wave tiles)
#define NP600 608                   // N=600 padded to 32 (19*32)
#define SCL_D 0.1414213562f         // 1/sqrt(50)

// ---------------- helpers ----------------
__device__ __forceinline__ unsigned short f2bf(float f) {
    unsigned int u = __float_as_uint(f);
    return (unsigned short)((u + 0x7fffu + ((u >> 16) & 1u)) >> 16);
}

__device__ __forceinline__ v16bf mk16(const unsigned short* p) {
    v8u a = *(const v8u*)p;
    v8u b = *(const v8u*)(p + 16);
    v16u t = __builtin_shufflevector(a, b, 0,1,2,3,4,5,6,7,8,9,10,11,12,13,14,15);
    return __builtin_bit_cast(v16bf, t);
}

template <int A> __device__ __forceinline__ float actf(float x) {
    if constexpr (A == 1) { return x > 0.f ? x : 0.01f * x; }                 // leaky
    else if constexpr (A == 2) {                                              // tanh-GELU
        float x3 = x * x * x;
        return 0.5f * x * (1.f + tanhf(0.79788456f * (x + 0.044715f * x3)));
    } else if constexpr (A == 3) { return 1.f / (1.f + expf(-x)); }           // sigmoid
    else if constexpr (A == 4) { return x > 0.f ? x : 0.f; }                  // relu
    else { return x; }
}

__device__ __forceinline__ void atomicMaxF(float* addr, float val) {
    if (val >= 0.f) atomicMax((int*)addr, __float_as_int(val));
    else            atomicMin((unsigned int*)addr, __float_as_uint(val));
}

// ================= WMMA GEMM, 32x32 tile per wave (2x2 register blocking) =================
// C(M,N) = act(A(M,Kp) @ W(Npad,Kp)^T * scale + bias);  M % 32 == 0, Npad % 32 == 0.
template <int ACT, bool ADD>
__global__ __launch_bounds__(32) void k_gemm(const unsigned short* __restrict__ A,
                                             const unsigned short* __restrict__ W,
                                             const float* __restrict__ bias,
                                             float* __restrict__ C,
                                             int M, int N, int Kp, float scale) {
    int tm = blockIdx.x, tn = blockIdx.y;
    int lane = threadIdx.x & 31;
    int half = lane >> 4, l15 = lane & 15;
    const unsigned short* ar0 = A + (size_t)(tm * 32 + l15)      * Kp + half * 8;
    const unsigned short* ar1 = A + (size_t)(tm * 32 + 16 + l15) * Kp + half * 8;
    const unsigned short* wr0 = W + (size_t)(tn * 32 + l15)      * Kp + half * 8;
    const unsigned short* wr1 = W + (size_t)(tn * 32 + 16 + l15) * Kp + half * 8;
    v8f a00 = {}, a01 = {}, a10 = {}, a11 = {};
    for (int k = 0; k < Kp; k += 32) {
        v16bf fa0 = mk16(ar0 + k);
        v16bf fa1 = mk16(ar1 + k);
        v16bf fb0 = mk16(wr0 + k);
        v16bf fb1 = mk16(wr1 + k);
        a00 = __builtin_amdgcn_wmma_f32_16x16x32_bf16(false, fa0, false, fb0, (short)0, a00, false, false);
        a01 = __builtin_amdgcn_wmma_f32_16x16x32_bf16(false, fa0, false, fb1, (short)0, a01, false, false);
        a10 = __builtin_amdgcn_wmma_f32_16x16x32_bf16(false, fa1, false, fb0, (short)0, a10, false, false);
        a11 = __builtin_amdgcn_wmma_f32_16x16x32_bf16(false, fa1, false, fb1, (short)0, a11, false, false);
    }
    v8f* accs[4] = {&a00, &a01, &a10, &a11};
#pragma unroll
    for (int mi = 0; mi < 2; ++mi) {
#pragma unroll
        for (int ni = 0; ni < 2; ++ni) {
            int col = tn * 32 + ni * 16 + l15;
            if (col >= N) continue;
            float bsv = bias ? bias[col] : 0.f;
            v8f& acc = *accs[mi * 2 + ni];
#pragma unroll
            for (int r = 0; r < 8; ++r) {
                int row = tm * 32 + mi * 16 + half * 8 + r;
                float v = actf<ACT>(acc[r] * scale + bsv);
                size_t o = (size_t)row * N + col;
                if (ADD) C[o] += v; else C[o] = v;
            }
        }
    }
}

// ================= fused triangle attention per (b, h, i) =================
__global__ __launch_bounds__(128) void k_tri_attn(const unsigned short* __restrict__ qp,
                                                  const unsigned short* __restrict__ kp,
                                                  const unsigned short* __restrict__ vp,
                                                  const float* __restrict__ biasb,
                                                  const float* __restrict__ gate,
                                                  unsigned short* __restrict__ outp) {
    int bhi = blockIdx.x;                       // ((b*H + h)*N + i)
    int i = bhi % NMAXc;
    int h = (bhi / NMAXc) % Hc;
    int b = bhi / (NMAXc * Hc);
    int wave = threadIdx.x >> 5;
    int lane = threadIdx.x & 31;
    int half = lane >> 4, l15 = lane & 15;
    int j0 = wave * 16;

    const unsigned short* qb = qp + (size_t)bhi * NMAXc * Dpc;  // rows j,  K = Dpad
    const unsigned short* kb = kp + (size_t)bhi * NMAXc * Dpc;  // rows k,  K = Dpad
    const unsigned short* vb = vp + (size_t)bhi * Dpc * NMAXc;  // rows d,  K = Nmax

    __shared__ unsigned short sprob[4][16 * 64];

    // ---- scores: S(16x64) = q(16xDp) @ k(64xDp)^T ----
    v8f sc[4] = {v8f{}, v8f{}, v8f{}, v8f{}};
    for (int kk = 0; kk < Dpc; kk += 32) {
        const unsigned short* ar = qb + (size_t)(j0 + l15) * Dpc + half * 8 + kk;
        v16bf av = mk16(ar);
#pragma unroll
        for (int nt = 0; nt < 4; ++nt) {
            const unsigned short* br = kb + (size_t)(nt * 16 + l15) * Dpc + half * 8 + kk;
            v16bf bv = mk16(br);
            sc[nt] = __builtin_amdgcn_wmma_f32_16x16x32_bf16(false, av, false, bv,
                                                             (short)0, sc[nt], false, false);
        }
    }
    // ---- scale + bias + row softmax ----
#pragma unroll
    for (int r = 0; r < 8; ++r) {
        int row = j0 + half * 8 + r;
        float bv_ = biasb[(((size_t)(b * NMAXc + i)) * NMAXc + row) * Hc + h];
        float m = -1e30f;
#pragma unroll
        for (int nt = 0; nt < 4; ++nt) {
            float s = sc[nt][r] * SCL_D + bv_;
            sc[nt][r] = s;
            m = fmaxf(m, s);
        }
        for (int msk = 1; msk < 16; msk <<= 1) m = fmaxf(m, __shfl_xor(m, msk, 32));
        float ssum = 0.f;
#pragma unroll
        for (int nt = 0; nt < 4; ++nt) {
            float e = expf(sc[nt][r] - m);
            sc[nt][r] = e;
            ssum += e;
        }
        for (int msk = 1; msk < 16; msk <<= 1) ssum += __shfl_xor(ssum, msk, 32);
        float inv = 1.f / ssum;
#pragma unroll
        for (int nt = 0; nt < 4; ++nt)
            sprob[wave][(half * 8 + r) * 64 + nt * 16 + l15] = f2bf(sc[nt][r] * inv);
    }
    // ---- out: O(16xDp) = P(16x64) @ V(Dp x 64)^T  (V packed as (d,k)) ----
    v8f oc[4] = {v8f{}, v8f{}, v8f{}, v8f{}};
    for (int kk = 0; kk < NMAXc; kk += 32) {
        const unsigned short* ar = &sprob[wave][l15 * 64 + half * 8 + kk];
        v16bf av = mk16(ar);
#pragma unroll
        for (int nt = 0; nt < 4; ++nt) {
            const unsigned short* br = vb + (size_t)(nt * 16 + l15) * NMAXc + half * 8 + kk;
            v16bf bv = mk16(br);
            oc[nt] = __builtin_amdgcn_wmma_f32_16x16x32_bf16(false, av, false, bv,
                                                             (short)0, oc[nt], false, false);
        }
    }
    // ---- gate + store into packed (RT x Gpad) bf16 buffer ----
#pragma unroll
    for (int nt = 0; nt < 4; ++nt) {
        int d = nt * 16 + l15;
        if (d < Dc) {
#pragma unroll
            for (int r = 0; r < 8; ++r) {
                int j = j0 + half * 8 + r;
                size_t row = ((size_t)(b * NMAXc + i)) * NMAXc + j;
                float gv = gate[row * Gc + h * Dc + d];
                outp[row * Gpc + h * Dc + d] = f2bf(gv * oc[nt][r]);
            }
        }
    }
}

// ================= small elementwise / graph kernels =================
__global__ void k_fill_f32(float* p, float v, long long n) {
    long long t = blockIdx.x * (long long)blockDim.x + threadIdx.x;
    if (t < n) p[t] = v;
}
__global__ void k_fill_u16(unsigned short* p, unsigned short v, long long n) {
    long long t = blockIdx.x * (long long)blockDim.x + threadIdx.x;
    if (t < n) p[t] = v;
}
__global__ void k_pack(const float* __restrict__ s, unsigned short* __restrict__ d,
                       int rows, int C, int Cp, long long n) {
    long long t = blockIdx.x * (long long)blockDim.x + threadIdx.x;
    if (t >= n) return;
    int c = (int)(t % Cp);
    long long r = t / Cp;
    d[t] = (r < rows && c < C) ? f2bf(s[(size_t)r * C + c]) : (unsigned short)0;
}
__global__ void k_pack_he1in(const float* __restrict__ nf, const int* __restrict__ srci,
                             const float* __restrict__ ef, unsigned short* __restrict__ d,
                             int E, long long n) {
    long long t = blockIdx.x * (long long)blockDim.x + threadIdx.x;
    if (t >= n) return;
    int c = (int)(t % KNFp);
    int e = (int)(t / KNFp);
    unsigned short o = 0;
    if (c < NFc)            o = f2bf(nf[(size_t)srci[e] * NFc + c]);
    else if (c < NFc + EFc) o = f2bf(ef[(size_t)e * EFc + (c - NFc)]);
    d[t] = o;
}
__global__ void k_edge_logits(const float* __restrict__ X1, const int* __restrict__ i1,
                              const float* __restrict__ X2, const int* __restrict__ i2,
                              const float* __restrict__ w, const float* __restrict__ bptr,
                              float* __restrict__ lg, int E, int Gd) {
    int e = blockIdx.x; int lane = threadIdx.x;
    if (e >= E) return;
    int r1 = i1 ? i1[e] : e;
    int r2 = i2 ? i2[e] : e;
    const float* x1 = X1 + (size_t)r1 * Gd;
    const float* x2 = X2 + (size_t)r2 * Gd;
    float s = 0.f;
    for (int g = lane; g < Gd; g += 32) s += x1[g] * w[g] + x2[g] * w[Gd + g];
    for (int m = 16; m; m >>= 1) s += __shfl_xor(s, m, 32);
    if (lane == 0) { s += bptr[0]; lg[e] = s > 0.f ? s : 0.01f * s; }
}
__global__ void k_segmax(const float* lg, const int* di, float* m, int E) {
    int t = blockIdx.x * blockDim.x + threadIdx.x;
    if (t < E) atomicMaxF(&m[di[t]], lg[t]);
}
__global__ void k_expsum(const float* lg, const int* di, const float* m,
                         float* e, float* s, int E) {
    int t = blockIdx.x * blockDim.x + threadIdx.x;
    if (t >= E) return;
    float v = expf(lg[t] - m[di[t]]);
    e[t] = v;
    atomicAdd(&s[di[t]], v);
}
__global__ void k_segnorm(const float* e, const int* di, const float* s, float* a, int E) {
    int t = blockIdx.x * blockDim.x + threadIdx.x;
    if (t < E) a[t] = e[t] / s[di[t]];
}
__global__ void k_scatter(const float* __restrict__ vals, const int* __restrict__ si,
                          const float* __restrict__ a, const int* __restrict__ di,
                          float* __restrict__ c, int E, int Gd) {
    long long t = blockIdx.x * (long long)blockDim.x + threadIdx.x;
    if (t >= (long long)E * Gd) return;
    int e = (int)(t / Gd), g = (int)(t % Gd);
    int sr = si ? si[e] : e;
    atomicAdd(&c[(size_t)di[e] * Gd + g], a[e] * vals[(size_t)sr * Gd + g]);
}
__global__ void k_elu(const float* x, float* y, long long n) {
    long long t = blockIdx.x * (long long)blockDim.x + threadIdx.x;
    if (t >= n) return;
    float v = x[t];
    y[t] = v > 0.f ? v : expm1f(v);
}
__global__ void k_gru(const float* __restrict__ gi, const float* __restrict__ gh,
                      const float* __restrict__ hprev, float* __restrict__ hout,
                      long long n) {
    long long t = blockIdx.x * (long long)blockDim.x + threadIdx.x;
    if (t >= n) return;
    long long v = t / Gc, g = t % Gc;
    size_t b3 = (size_t)v * G3c;
    float ir = gi[b3 + g], iz = gi[b3 + Gc + g], in_ = gi[b3 + 2 * Gc + g];
    float hr = gh[b3 + g], hz = gh[b3 + Gc + g], hn  = gh[b3 + 2 * Gc + g];
    float r = 1.f / (1.f + expf(-(ir + hr)));
    float z = 1.f / (1.f + expf(-(iz + hz)));
    float nn = tanhf(in_ + r * hn);
    float o = (1.f - z) * nn + z * hprev[t];
    hout[t] = o > 0.f ? o : 0.f;
}
__global__ __launch_bounds__(128) void k_ln(const float* __restrict__ x,
                                            const float* __restrict__ w,
                                            const float* __restrict__ b,
                                            float* __restrict__ y, int C, float eps) {
    int row = blockIdx.x, tid = threadIdx.x;
    const float* xr = x + (size_t)row * C;
    float s = 0.f, s2 = 0.f;
    for (int c = tid; c < C; c += 128) { float v = xr[c]; s += v; s2 += v * v; }
    __shared__ float sh[128], sh2[128];
    sh[tid] = s; sh2[tid] = s2; __syncthreads();
    for (int st = 64; st > 0; st >>= 1) {
        if (tid < st) { sh[tid] += sh[tid + st]; sh2[tid] += sh2[tid + st]; }
        __syncthreads();
    }
    float m = sh[0] / C;
    float inv = rsqrtf(sh2[0] / C - m * m + eps);
    float* yr = y + (size_t)row * C;
    for (int c = tid; c < C; c += 128) yr[c] = (xr[c] - m) * inv * w[c] + b[c];
}
__global__ void k_rowdotH(const float* __restrict__ x, const float* __restrict__ w,
                          float* __restrict__ o, int Gd) {
    int row = blockIdx.x, lane = threadIdx.x;
    const float* xr = x + (size_t)row * Gd;
    for (int h = 0; h < Hc; ++h) {
        float s = 0.f;
        for (int g = lane; g < Gd; g += 32) s += xr[g] * w[h * Gd + g];
        for (int m = 16; m; m >>= 1) s += __shfl_xor(s, m, 32);
        if (lane == 0) o[(size_t)row * Hc + h] = s;
    }
}
__global__ void k_mean_rows(const float* __restrict__ x, float* __restrict__ y,
                            int Bn, int R, int Gd) {
    int t = blockIdx.x * blockDim.x + threadIdx.x;
    if (t >= Bn * Gd) return;
    int b = t / Gd, g = t % Gd;
    float s = 0.f;
    for (int r = 0; r < R; ++r) s += x[((size_t)b * R + r) * Gd + g];
    y[t] = s / (float)R;
}
__global__ void k_hb(const float* h, const float* pgm, float* hB, long long n) {
    long long t = blockIdx.x * (long long)blockDim.x + threadIdx.x;
    if (t >= n) return;
    long long v = t / Gc, g = t % Gc;
    hB[t] = h[t] + pgm[(v / NMAXc) * Gc + g];
}
__global__ void k_pairk(const float* __restrict__ hB, float* __restrict__ pair, long long n) {
    long long t = blockIdx.x * (long long)blockDim.x + threadIdx.x;
    if (t >= n) return;
    int g = (int)(t % Gc);
    long long r = t / Gc;
    int j = (int)(r % NMAXc); r /= NMAXc;
    int i = (int)(r % NMAXc);
    int b = (int)(r / NMAXc);
    pair[t] = hB[((size_t)(b * NMAXc + i)) * Gc + g] + hB[((size_t)(b * NMAXc + j)) * Gc + g];
}
__global__ void k_pack_qk(const float* __restrict__ t, unsigned short* __restrict__ d,
                          long long n) {
    long long idx = blockIdx.x * (long long)blockDim.x + threadIdx.x;
    if (idx >= n) return;
    int dd = (int)(idx % Dpc);
    long long r = idx / Dpc;
    int j = (int)(r % NMAXc); r /= NMAXc;
    int i = (int)(r % NMAXc); r /= NMAXc;
    int h = (int)(r % Hc);
    int b = (int)(r / Hc);
    unsigned short v = 0;
    if (dd < Dc)
        v = f2bf(t[(((size_t)(b * NMAXc + i)) * NMAXc + j) * Gc + h * Dc + dd]);
    d[idx] = v;
}
__global__ void k_pack_vT(const float* __restrict__ t, unsigned short* __restrict__ d,
                          long long n) {
    long long idx = blockIdx.x * (long long)blockDim.x + threadIdx.x;
    if (idx >= n) return;
    int kk = (int)(idx % NMAXc);
    long long r = idx / NMAXc;
    int dd = (int)(r % Dpc); r /= Dpc;
    int i = (int)(r % NMAXc); r /= NMAXc;
    int h = (int)(r % Hc);
    int b = (int)(r / Hc);
    unsigned short v = 0;
    if (dd < Dc)
        v = f2bf(t[(((size_t)(b * NMAXc + i)) * NMAXc + kk) * Gc + h * Dc + dd]);
    d[idx] = v;
}
__global__ void k_pred(const float* __restrict__ graph, const float* __restrict__ w,
                       const float* __restrict__ bptr, float* __restrict__ out) {
    int b = blockIdx.x, lane = threadIdx.x;
    const float* xr = graph + (size_t)b * Gc;
    float s = 0.f;
    for (int g = lane; g < Gc; g += 32) s += xr[g] * w[g];
    for (int m = 16; m; m >>= 1) s += __shfl_xor(s, m, 32);
    if (lane == 0) out[b] = 1.f / (1.f + expf(-(s + bptr[0])));
}

// ================= host side =================
static inline long long cdivll(long long a, long long b) { return (a + b - 1) / b; }

extern "C" void kernel_launch(void* const* d_in, const int* in_sizes, int n_in,
                              void* d_out, int out_size, void* d_ws, size_t ws_size,
                              hipStream_t stream) {
    (void)out_size; (void)ws_size; (void)n_in;
    const float* node_feats    = (const float*)d_in[0];
    const float* edge_feats    = (const float*)d_in[1];
    const float* protein_feats = (const float*)d_in[2];
    const int*   src   = (const int*)d_in[3];
    const int*   dst   = (const int*)d_in[4];
    const int*   src_c = (const int*)d_in[5];
    const int*   dst_c = (const int*)d_in[6];
    const int E1 = in_sizes[3];     // 6048  (multiple of 32)
    const int EC = in_sizes[5];     // 193536

    // params (dict order)
    enum {
        GC_PN_W = 0, GC_PN_B, GC_PE1_W, GC_PE1_B, GC_PE2_W, GC_PE2_B, GC_ET_W, GC_ET_B,
        GC_WIH, GC_WHH, GC_BIH, GC_BHH,
        PFF_LN_W, PFF_LN_B, PFF_W1, PFF_B1, PFF_W2, PFF_B2, PFC_W, PFC_B,
        TA_LN_W, TA_LN_B, TA_Q, TA_K, TA_V, TA_B, TA_G_W, TA_G_B, TA_OUT_W, TA_OUT_B,
        PRED_W, PRED_B, LBASE
    };
    auto P = [&](int i) { return (const float*)d_in[7 + i]; };

    // ---- bump allocator over d_ws ----
    size_t off = 0;
    auto alloc = [&](size_t elems, size_t esz) -> void* {
        off = (off + 255) & ~(size_t)255;
        void* p = (char*)d_ws + off;
        off += elems * esz;
        return p;
    };
    auto af = [&](size_t n) { return (float*)alloc(n, 4); };
    auto au = [&](size_t n) { return (unsigned short*)alloc(n, 2); };

    // packed weights (bf16, K padded to 32, N padded to 32 rows)
    unsigned short* wp_gc_pn  = au((size_t)NP200 * KNFp);
    unsigned short* wp_gc_pe1 = au((size_t)NP200 * KNFp);
    unsigned short* wp_gc_et  = au((size_t)NP200 * Gpc);
    unsigned short* wp_gc_wih = au((size_t)NP600 * Gpc);
    unsigned short* wp_gc_whh = au((size_t)NP600 * Gpc);
    unsigned short* wp_pn[3], *wp_wih[3], *wp_whh[3];
    for (int i = 0; i < 3; ++i) {
        wp_pn[i]  = au((size_t)NP200 * Gpc);
        wp_wih[i] = au((size_t)NP600 * Gpc);
        wp_whh[i] = au((size_t)NP600 * Gpc);
    }
    unsigned short* wp_ff1 = au((size_t)2560 * 1280);
    unsigned short* wp_ff2 = au((size_t)256 * 2560);
    unsigned short* wp_pfc = au((size_t)NP200 * 256);
    unsigned short* wp_q   = au((size_t)NP200 * Gpc);
    unsigned short* wp_k   = au((size_t)NP200 * Gpc);
    unsigned short* wp_v   = au((size_t)NP200 * Gpc);
    unsigned short* wp_g   = au((size_t)NP200 * Gpc);
    unsigned short* wp_o   = au((size_t)NP200 * Gpc);

    // graph-stage buffers
    float* hv_new = af((size_t)Vc * Gc);
    float* he1    = af((size_t)E1 * Gc);
    float* et     = af((size_t)E1 * Gc);
    float* logits = af(EC);
    float* ebuf   = af(EC);
    float* abuf   = af(EC);
    float* mbuf   = af(Vc);
    float* sbuf   = af(Vc);
    float* cbuf   = af((size_t)Vc * Gc);
    float* xg     = af((size_t)Vc * Gc);
    float* gi     = af((size_t)Vc * G3c);
    float* gh     = af((size_t)Vc * G3c);
    float* hbuf   = af((size_t)Vc * Gc);
    float* hproj  = af((size_t)Vc * Gc);
    float* pgm    = af((size_t)Bc * Gc);
    float* graphb = af((size_t)Bc * Gc);
    float* biasb  = af((size_t)RTc * Hc);

    unsigned short* nf_p     = au((size_t)Vc * KNFp);
    unsigned short* he1in_p  = au((size_t)E1 * KNFp);
    unsigned short* he1_p    = au((size_t)E1 * Gpc);
    unsigned short* xg_p     = au((size_t)Vc * Gpc);
    unsigned short* h_p      = au((size_t)Vc * Gpc);
    unsigned short* px_p     = au((size_t)RPc * 1280);
    unsigned short* ff1_p    = au((size_t)RPc * 2560);
    unsigned short* ff2_p    = au((size_t)RPc * 256);
    unsigned short* xln_p    = au((size_t)RTc * Gpc);
    unsigned short* outg_p   = au((size_t)RTc * Gpc);
    unsigned short* qp       = au((size_t)Bc * Hc * NMAXc * NMAXc * Dpc);
    unsigned short* kp       = au((size_t)Bc * Hc * NMAXc * NMAXc * Dpc);
    unsigned short* vp       = au((size_t)Bc * Hc * NMAXc * NMAXc * Dpc);

    // big reusable f32 arenas
    float* arenaA = af((size_t)RPc * 2560);    // px -> ff1 -> ff2 -> pg
    float* pair   = af((size_t)RTc * Gc);
    float* xln_f  = af((size_t)RTc * Gc);
    float* gate_f = af((size_t)RTc * Gc);
    float* tmp_f  = af((size_t)RTc * Gc);      // q/k/v f32 scratch

    auto packW = [&](const float* w, unsigned short* dp, int rows, int rowsPad, int C, int Cp) {
        long long n = (long long)rowsPad * Cp;
        k_pack<<<cdivll(n, 256), 256, 0, stream>>>(w, dp, rows, C, Cp, n);
    };
    auto packA = [&](const float* w, unsigned short* dp, int rows, int C, int Cp) {
        packW(w, dp, rows, rows, C, Cp);
    };

    // ---------------- pack weights ----------------
    packW(P(GC_PN_W),  wp_gc_pn,  Gc, NP200, NFc, KNFp);
    packW(P(GC_PE1_W), wp_gc_pe1, Gc, NP200, NFc + EFc, KNFp);
    packW(P(GC_ET_W),  wp_gc_et,  Gc, NP200, Gc, Gpc);
    packW(P(GC_WIH),   wp_gc_wih, G3c, NP600, Gc, Gpc);
    packW(P(GC_WHH),   wp_gc_whh, G3c, NP600, Gc, Gpc);
    for (int i = 0; i < 3; ++i) {
        packW(P(LBASE + 8 * i + 2), wp_pn[i],  Gc,  NP200, Gc, Gpc);
        packW(P(LBASE + 8 * i + 4), wp_wih[i], G3c, NP600, Gc, Gpc);
        packW(P(LBASE + 8 * i + 5), wp_whh[i], G3c, NP600, Gc, Gpc);
    }
    packW(P(PFF_W1), wp_ff1, 2560, 2560, 1280, 1280);
    packW(P(PFF_W2), wp_ff2, 256, 256, 2560, 2560);
    packW(P(PFC_W),  wp_pfc, Gc, NP200, 256, 256);
    packW(P(TA_Q),     wp_q, Gc, NP200, Gc, Gpc);
    packW(P(TA_K),     wp_k, Gc, NP200, Gc, Gpc);
    packW(P(TA_V),     wp_v, Gc, NP200, Gc, Gpc);
    packW(P(TA_G_W),   wp_g, Gc, NP200, Gc, Gpc);
    packW(P(TA_OUT_W), wp_o, Gc, NP200, Gc, Gpc);

    // 32x32 wave tiles: grid = (M/32, Npad/32)
    dim3 g200(Vc / 32, NP200 / 32), g600(Vc / 32, NP600 / 32);

    // ---------------- GetContext ----------------
    packA(node_feats, nf_p, Vc, NFc, KNFp);
    k_gemm<1, false><<<g200, 32, 0, stream>>>(nf_p, wp_gc_pn, P(GC_PN_B), hv_new, Vc, Gc, KNFp, 1.f);
    {
        long long n = (long long)E1 * KNFp;
        k_pack_he1in<<<cdivll(n, 256), 256, 0, stream>>>(node_feats, src, edge_feats, he1in_p, E1, n);
    }
    k_gemm<1, false><<<dim3(E1 / 32, NP200 / 32), 32, 0, stream>>>(he1in_p, wp_gc_pe1, P(GC_PE1_B),
                                                                   he1, E1, Gc, KNFp, 1.f);
    k_edge_logits<<<E1, 32, 0, stream>>>(hv_new, dst, he1, nullptr, P(GC_PE2_W), P(GC_PE2_B),
                                         logits, E1, Gc);
    auto edge_softmax = [&](const int* di, int E) {
        k_fill_f32<<<cdivll(Vc, 256), 256, 0, stream>>>(mbuf, -1e30f, Vc);
        k_fill_f32<<<cdivll(Vc, 256), 256, 0, stream>>>(sbuf, 0.f, Vc);
        k_segmax<<<cdivll(E, 256), 256, 0, stream>>>(logits, di, mbuf, E);
        k_expsum<<<cdivll(E, 256), 256, 0, stream>>>(logits, di, mbuf, ebuf, sbuf, E);
        k_segnorm<<<cdivll(E, 256), 256, 0, stream>>>(ebuf, di, sbuf, abuf, E);
    };
    edge_softmax(dst, E1);
    packA(he1, he1_p, E1, Gc, Gpc);
    k_gemm<0, false><<<dim3(E1 / 32, NP200 / 32), 32, 0, stream>>>(he1_p, wp_gc_et, P(GC_ET_B),
                                                                   et, E1, Gc, Gpc, 1.f);
    k_fill_f32<<<cdivll((long long)Vc * Gc, 256), 256, 0, stream>>>(cbuf, 0.f, (long long)Vc * Gc);
    k_scatter<<<cdivll((long long)E1 * Gc, 256), 256, 0, stream>>>(et, nullptr, abuf, dst, cbuf, E1, Gc);
    auto gru_step = [&](const unsigned short* wih, const unsigned short* whh,
                        const float* bih, const float* bhh, const float* hprev) {
        k_elu<<<cdivll((long long)Vc * Gc, 256), 256, 0, stream>>>(cbuf, xg, (long long)Vc * Gc);
        packA(xg, xg_p, Vc, Gc, Gpc);
        packA(hprev, h_p, Vc, Gc, Gpc);
        k_gemm<0, false><<<g600, 32, 0, stream>>>(xg_p, wih, bih, gi, Vc, G3c, Gpc, 1.f);
        k_gemm<0, false><<<g600, 32, 0, stream>>>(h_p,  whh, bhh, gh, Vc, G3c, Gpc, 1.f);
        k_gru<<<cdivll((long long)Vc * Gc, 256), 256, 0, stream>>>(gi, gh, hprev, hbuf,
                                                                   (long long)Vc * Gc);
    };
    gru_step(wp_gc_wih, wp_gc_whh, P(GC_BIH), P(GC_BHH), hv_new);

    // ---------------- 3 GNN layers ----------------
    for (int i = 0; i < 3; ++i) {
        const int* s_ = (i < 2) ? src : src_c;
        const int* d_ = (i < 2) ? dst : dst_c;
        int E = (i < 2) ? E1 : EC;
        k_edge_logits<<<E, 32, 0, stream>>>(hbuf, d_, hbuf, s_, P(LBASE + 8 * i + 0),
                                            P(LBASE + 8 * i + 1), logits, E, Gc);
        edge_softmax(d_, E);
        packA(hbuf, h_p, Vc, Gc, Gpc);
        k_gemm<0, false><<<g200, 32, 0, stream>>>(h_p, wp_pn[i], P(LBASE + 8 * i + 3),
                                                  hproj, Vc, Gc, Gpc, 1.f);
        k_fill_f32<<<cdivll((long long)Vc * Gc, 256), 256, 0, stream>>>(cbuf, 0.f, (long long)Vc * Gc);
        k_scatter<<<cdivll((long long)E * Gc, 256), 256, 0, stream>>>(hproj, s_, abuf, d_, cbuf, E, Gc);
        gru_step(wp_wih[i], wp_whh[i], P(LBASE + 8 * i + 6), P(LBASE + 8 * i + 7), hbuf);
    }

    // ---------------- protein branch ----------------
    k_ln<<<RPc, 128, 0, stream>>>(protein_feats, P(PFF_LN_W), P(PFF_LN_B), arenaA, 1280, 1e-6f);
    packA(arenaA, px_p, RPc, 1280, 1280);
    k_gemm<2, false><<<dim3(RPc / 32, 2560 / 32), 32, 0, stream>>>(px_p, wp_ff1, P(PFF_B1),
                                                                   arenaA, RPc, 2560, 1280, 1.f);
    packA(arenaA, ff1_p, RPc, 2560, 2560);
    k_gemm<0, false><<<dim3(RPc / 32, 256 / 32), 32, 0, stream>>>(ff1_p, wp_ff2, P(PFF_B2),
                                                                  arenaA, RPc, 256, 2560, 1.f);
    packA(arenaA, ff2_p, RPc, 256, 256);
    k_gemm<0, false><<<dim3(RPc / 32, NP200 / 32), 32, 0, stream>>>(ff2_p, wp_pfc, P(PFC_B),
                                                                    arenaA, RPc, Gc, 256, 1.f);
    k_mean_rows<<<cdivll((long long)Bc * Gc, 256), 256, 0, stream>>>(arenaA, pgm, Bc, Lc, Gc);

    // ---------------- fuse + triangle attention ----------------
    k_hb<<<cdivll((long long)Vc * Gc, 256), 256, 0, stream>>>(hbuf, pgm, hproj, (long long)Vc * Gc);
    k_pairk<<<cdivll((long long)RTc * Gc, 256), 256, 0, stream>>>(hproj, pair, (long long)RTc * Gc);
    k_ln<<<RTc, 128, 0, stream>>>(pair, P(TA_LN_W), P(TA_LN_B), xln_f, Gc, 1e-5f);
    packA(xln_f, xln_p, RTc, Gc, Gpc);
    k_rowdotH<<<RTc, 32, 0, stream>>>(xln_f, P(TA_B), biasb, Gc);
    dim3 gRT(RTc / 32, NP200 / 32);
    k_gemm<3, false><<<gRT, 32, 0, stream>>>(xln_p, wp_g, P(TA_G_B), gate_f, RTc, Gc, Gpc, 1.f);
    long long nqk = (long long)Bc * Hc * NMAXc * NMAXc * Dpc;
    k_gemm<0, false><<<gRT, 32, 0, stream>>>(xln_p, wp_q, nullptr, tmp_f, RTc, Gc, Gpc, SCL_D);
    k_pack_qk<<<cdivll(nqk, 256), 256, 0, stream>>>(tmp_f, qp, nqk);
    k_gemm<0, false><<<gRT, 32, 0, stream>>>(xln_p, wp_k, nullptr, tmp_f, RTc, Gc, Gpc, 1.f);
    k_pack_qk<<<cdivll(nqk, 256), 256, 0, stream>>>(tmp_f, kp, nqk);
    k_gemm<0, false><<<gRT, 32, 0, stream>>>(xln_p, wp_v, nullptr, tmp_f, RTc, Gc, Gpc, 1.f);
    k_pack_vT<<<cdivll(nqk, 256), 256, 0, stream>>>(tmp_f, vp, nqk);
    k_fill_u16<<<cdivll((long long)RTc * Gpc, 256), 256, 0, stream>>>(outg_p, 0,
                                                                      (long long)RTc * Gpc);
    k_tri_attn<<<Bc * Hc * NMAXc, 128, 0, stream>>>(qp, kp, vp, biasb, gate_f, outg_p);
    k_gemm<0, true><<<gRT, 32, 0, stream>>>(outg_p, wp_o, P(TA_OUT_B), pair, RTc, Gc, Gpc, 1.f);

    // ---------------- pooling + prediction ----------------
    k_mean_rows<<<cdivll((long long)Bc * Gc, 256), 256, 0, stream>>>(pair, graphb, Bc,
                                                                     NMAXc * NMAXc, Gc);
    k_pred<<<Bc, 32, 0, stream>>>(graphb, P(PRED_W), P(PRED_B), (float*)d_out);
}
